// DecoderCRF_858993459733
// MI455X (gfx1250) — matrix operand order repaired and checked by reading, not compiled, not run on hardware
//
#include <hip/hip_runtime.h>
#include <hip/hip_bf16.h>

#define SEQ    16384
#define DIM    2048
#define KTAG   50
#define KP     64          // padded tag dimension
#define START_ 48
#define STOP_  49
#define NEGV   (-10000.0f)
#define CHUNK  64
#define NCHUNK 256

typedef __attribute__((ext_vector_type(16))) __bf16 v16bf;
typedef __attribute__((ext_vector_type(8)))  float  v8f;
typedef __attribute__((ext_vector_type(2)))  float  v2f;
typedef __attribute__((ext_vector_type(4)))  unsigned int v4u;
typedef __attribute__((ext_vector_type(8)))  int    v8i;
typedef __attribute__((ext_vector_type(4)))  int    v4i;

#if defined(__AMDGCN__) && __has_builtin(__builtin_amdgcn_wmma_f32_16x16x4_f32)
#define HAVE_WMMA_F32X4 1
#else
#define HAVE_WMMA_F32X4 0
#if defined(__AMDGCN__)
#warning "wmma_f32_16x16x4_f32 builtin not available; chunk kernel using scalar fallback"
#endif
#endif

#if defined(__AMDGCN__) && __has_builtin(__builtin_amdgcn_tensor_load_to_lds) && \
    __has_builtin(__builtin_amdgcn_s_wait_tensorcnt)
#define HAVE_TDM 1
#else
#define HAVE_TDM 0
#if defined(__AMDGCN__)
#warning "tensor_load_to_lds builtin not available; using plain LDS staging"
#endif
#endif

// ---------------------------------------------------------------------------
// TDM helper: DMA a 2D f32 tile (rows x dim0, line stride stride0 elements)
// from global memory into LDS at byte offset lds_off. Descriptor per ISA
// cdna5 §8 (D# group0/group1); groups 2/3 zero (<=2D tensor). Issue from ONE
// wave only; caller must s_wait_tensorcnt + barrier before consuming.
// ---------------------------------------------------------------------------
#if HAVE_TDM
__device__ __forceinline__ int rfl_(unsigned int x) {
  return __builtin_amdgcn_readfirstlane((int)x);
}
__device__ __forceinline__ void tdm_load_f32(unsigned int lds_off, const void* gptr,
                                             unsigned int dim0, unsigned int rows,
                                             unsigned int stride0) {
  const unsigned long long ga = (unsigned long long)gptr;
  v4u g0;
  g0[0] = (unsigned int)rfl_(1u);                          // count=1, user mode
  g0[1] = (unsigned int)rfl_(lds_off);                     // LDS byte address
  g0[2] = (unsigned int)rfl_((unsigned int)ga);            // global_addr[31:0]
  g0[3] = (unsigned int)rfl_((unsigned int)((ga >> 32) & 0x1FFFFFFu) | (2u << 30)); // [56:32]|type=2
  v8i g1;
  g1[0] = rfl_(2u << 16);                                  // data_size=4B; wg_mask=0; no flags
  g1[1] = rfl_(dim0 << 16);                                // tensor_dim0[15:0] (hi half)
  g1[2] = rfl_((dim0 >> 16) | (rows << 16));               // tensor_dim0[31:16] | tensor_dim1[15:0]
  g1[3] = rfl_((rows >> 16) | (dim0 << 16));               // tensor_dim1[31:16] | tile_dim0
  g1[4] = rfl_(rows & 0xFFFFu);                            // tile_dim1 | tile_dim2=0
  g1[5] = rfl_(stride0);                                   // tensor_dim0_stride[31:0]
  g1[6] = 0;                                               // stride0[47:32] | stride1[15:0]
  g1[7] = 0;
  const v4i gz = {0, 0, 0, 0};
#if __clang_major__ >= 23
  const v8i gz8 = {0, 0, 0, 0, 0, 0, 0, 0};
  __builtin_amdgcn_tensor_load_to_lds(g0, g1, gz, gz, gz8, 0);
#else
  __builtin_amdgcn_tensor_load_to_lds(g0, g1, gz, gz, 0);
#endif
}
#endif

// ---------------------------------------------------------------------------
// Kernel 1: feats[t, k] = X[t,:] . W[k,:] + b[k]   (k<50; k>=50 -> NEG pad)
// HBM-bound (134 MB of X at 23.3 TB/s ~ 5.8us): bf16 WMMA, branch-free k-loop.
// ---------------------------------------------------------------------------
__global__ __launch_bounds__(128) void feats_gemm(const float* __restrict__ X,
                                                  const float* __restrict__ W,
                                                  const float* __restrict__ bias,
                                                  float* __restrict__ feats) {
  const int wave = threadIdx.x >> 5;
  const int lane = threadIdx.x & 31;
  const int half = lane >> 4;
  const int lm   = lane & 15;
  const int rowBase = blockIdx.x * 64 + wave * 16;
  const int arow = rowBase + lm;

  // Hoisted per-tile W row pointers + zero-masks (kills exec divergence in loop)
  const float* wp[4];
  float wmask[4];
#pragma unroll
  for (int t = 0; t < 4; ++t) {
    const int wrow = t * 16 + lm;                      // output column == W row
    const int wr   = (wrow < KTAG) ? wrow : (KTAG - 1);
    wmask[t] = (wrow < KTAG) ? 1.0f : 0.0f;
    wp[t] = W + (size_t)wr * DIM + 8 * half;
  }
  const float* ap = X + (size_t)arow * DIM + 8 * half;

  v8f acc[4] = {};
  for (int k0 = 0; k0 < DIM; k0 += 32) {
    // A fragment (bf16 16x32): lane (lm,half) holds row arow,
    // K = 8*half + {0..7}  and  K = 16 + 8*half + {0..7}
    float av[16];
    *(float4*)&av[0]  = ((const float4*)(ap + k0))[0];
    *(float4*)&av[4]  = ((const float4*)(ap + k0))[1];
    *(float4*)&av[8]  = ((const float4*)(ap + k0))[4];   // +16 floats
    *(float4*)&av[12] = ((const float4*)(ap + k0))[5];
    v16bf a;
#pragma unroll
    for (int i = 0; i < 16; ++i) a[i] = (__bf16)av[i];

#pragma unroll
    for (int t = 0; t < 4; ++t) {
      // B fragment (bf16 32x16): lane holds col (t*16+lm), K = 8*half + {0..7} and 16 + 8*half + {0..7}
      float bv[16];
      *(float4*)&bv[0]  = ((const float4*)(wp[t] + k0))[0];
      *(float4*)&bv[4]  = ((const float4*)(wp[t] + k0))[1];
      *(float4*)&bv[8]  = ((const float4*)(wp[t] + k0))[4];
      *(float4*)&bv[12] = ((const float4*)(wp[t] + k0))[5];
      v16bf bf;
#pragma unroll
      for (int i = 0; i < 16; ++i) bf[i] = (__bf16)(bv[i] * wmask[t]);
      acc[t] = __builtin_amdgcn_wmma_f32_16x16x32_bf16(
          false, a, false, bf, (short)0, acc[t], false, false);
    }
  }
  // D layout: VGPR r -> M = r + 8*half, N = lm
#pragma unroll
  for (int t = 0; t < 4; ++t) {
#pragma unroll
    for (int r = 0; r < 8; ++r) {
      const int row = rowBase + r + 8 * half;
      const int col = t * 16 + lm;
      const float v = (col < KTAG) ? (acc[t][r] + bias[col]) : NEGV;
      feats[(size_t)row * KP + col] = v;
    }
  }
}

// ---------------------------------------------------------------------------
// Kernel 2: per-chunk transfer matrix in the exp domain.
// P'[i,k] = exp(feat_t[i]) * sum_j E[i,j] * P[j,k],  E = exp(transitions),
// per-column offsets M_k rescaled every 4 steps. P transposed in LDS (stride
// 68 vs bank conflicts). E A-fragments register-resident for all 64 steps.
// feats block staged to LDS by the Tensor Data Mover.
// ---------------------------------------------------------------------------
__global__ __launch_bounds__(128) void crf_chunk(const float* __restrict__ feats,
                                                 const float* __restrict__ trans,
                                                 float* __restrict__ bufA) {
  __shared__ float PT[64][68];
  __shared__ float FE[64][64];
  __shared__ float Moff[64];
  const int tid  = threadIdx.x;
  const int wave = tid >> 5;
  const int lane = tid & 31;
  const int half = lane >> 4;
  const int lm   = lane & 15;
  const int c0   = blockIdx.x * CHUNK;
  const int irow = 16 * wave + lm;

#if HAVE_TDM
  if (tid < 32) {   // wave 0 issues one TDM op for the whole 16KB feats block
    tdm_load_f32((unsigned int)(uintptr_t)&FE[0][0], feats + (size_t)c0 * KP,
                 4096, 1, 4096);
    __builtin_amdgcn_s_wait_tensorcnt(0);
  }
#else
  for (int idx = tid; idx < 64 * 64; idx += 128)
    FE[idx >> 6][idx & 63] = feats[(size_t)(c0 + (idx >> 6)) * KP + (idx & 63)];
#endif
  for (int idx = tid; idx < 64 * 64; idx += 128) {
    const int k2 = idx >> 6, i2 = idx & 63;
    PT[k2][i2] = (k2 == i2) ? 1.0f : 0.0f;     // P = I (exp-domain identity)
  }
  if (tid < 64) Moff[tid] = 0.0f;

#if HAVE_WMMA_F32X4
  // A fragments (f32 16x4): lane row irow; kstep ks -> K = 4ks + 2*half + {0,1}
  float ea0[16], ea1[16];
#pragma unroll
  for (int ks = 0; ks < 16; ++ks) {
    const int j0 = 4 * ks + 2 * half;
    float e0 = 0.0f, e1 = 0.0f;
    if (irow < KTAG) {
      if (j0 < KTAG)     e0 = __expf(trans[irow * KTAG + j0]);
      if (j0 + 1 < KTAG) e1 = __expf(trans[irow * KTAG + j0 + 1]);
    }
    ea0[ks] = e0; ea1[ks] = e1;
  }
  __syncthreads();

  for (int s = 0; s < CHUNK; ++s) {
    const float ef = __expf(FE[s][irow]);      // fold diag(exp(feat)) into A rows
    v8f acc[4] = {};
#pragma unroll
    for (int ks = 0; ks < 16; ++ks) {
      v2f a;
      a.x = ea0[ks] * ef;
      a.y = ea1[ks] * ef;
      const int jb = 4 * ks + 2 * half;
#pragma unroll
      for (int t = 0; t < 4; ++t) {
        v2f b;
        b.x = PT[t * 16 + lm][jb];
        b.y = PT[t * 16 + lm][jb + 1];
        acc[t] = __builtin_amdgcn_wmma_f32_16x16x4_f32(
            false, a, false, b, (short)0, acc[t], false, false);
      }
    }
    __syncthreads();                            // all reads of P_s done
#pragma unroll
    for (int t = 0; t < 4; ++t)
#pragma unroll
      for (int r = 0; r < 8; ++r)
        PT[t * 16 + lm][16 * wave + 8 * half + r] = acc[t][r];
    __syncthreads();
    if ((s & 3) == 3) {                         // renormalize columns
      if (tid < 64) {
        float m = 1e-30f;
        for (int i2 = 0; i2 < 64; ++i2) m = fmaxf(m, PT[tid][i2]);
        const float inv = 1.0f / m;
        for (int i2 = 0; i2 < 64; ++i2) PT[tid][i2] *= inv;
        Moff[tid] += __logf(m);
      }
      __syncthreads();
    }
  }
#else
  // Scalar fallback (no f32 WMMA builtin on this toolchain)
  __shared__ float Esh[64][68];
  for (int idx = tid; idx < 64 * 64; idx += 128) {
    const int i2 = idx >> 6, j2 = idx & 63;
    Esh[i2][j2] = (i2 < KTAG && j2 < KTAG) ? __expf(trans[i2 * KTAG + j2]) : 0.0f;
  }
  __syncthreads();
  for (int s = 0; s < CHUNK; ++s) {
    float nv[32];
    for (int q = 0; q < 32; ++q) {
      const int idx = tid + q * 128;
      const int i2 = idx >> 6, k2 = idx & 63;
      float sum = 0.0f;
      for (int j = 0; j < 64; ++j) sum += Esh[i2][j] * PT[k2][j];
      nv[q] = sum * __expf(FE[s][i2]);
    }
    __syncthreads();
    for (int q = 0; q < 32; ++q) {
      const int idx = tid + q * 128;
      PT[idx & 63][idx >> 6] = nv[q];
    }
    __syncthreads();
    if ((s & 3) == 3) {
      if (tid < 64) {
        float m = 1e-30f;
        for (int i2 = 0; i2 < 64; ++i2) m = fmaxf(m, PT[tid][i2]);
        const float inv = 1.0f / m;
        for (int i2 = 0; i2 < 64; ++i2) PT[tid][i2] *= inv;
        Moff[tid] += __logf(m);
      }
      __syncthreads();
    }
  }
#endif

  float* out = bufA + (size_t)blockIdx.x * 4096;
  for (int idx = tid; idx < 4096; idx += 128) {
    const int i2 = idx >> 6, k2 = idx & 63;
    out[idx] = __logf(fmaxf(PT[k2][i2], 1e-38f)) + Moff[k2];
  }
}

// ---------------------------------------------------------------------------
// Kernel 3: log-semiring matmul pair-combine: dst[p] = src[2p+1] (later) o src[2p]
// C[i,k] = logsumexp_j ( A2[i,j] + A1[j,k] ).  Operand tiles staged via TDM.
// ---------------------------------------------------------------------------
__global__ __launch_bounds__(256) void lse_combine(const float* __restrict__ src,
                                                   float* __restrict__ dst) {
  __shared__ float A1[64][64];
  __shared__ float A2[64][64];
  const float* a1 = src + (size_t)(2 * blockIdx.x) * 4096;
  const float* a2 = a1 + 4096;
#if HAVE_TDM
  if (threadIdx.x < 32) {
    tdm_load_f32((unsigned int)(uintptr_t)&A1[0][0], a1, 4096, 1, 4096);
    tdm_load_f32((unsigned int)(uintptr_t)&A2[0][0], a2, 4096, 1, 4096);
    __builtin_amdgcn_s_wait_tensorcnt(0);
  }
#else
  for (int idx = threadIdx.x; idx < 4096; idx += 256) {
    ((float*)A1)[idx] = a1[idx];
    ((float*)A2)[idx] = a2[idx];
  }
#endif
  __syncthreads();
  float* out = dst + (size_t)blockIdx.x * 4096;
  for (int idx = threadIdx.x; idx < 4096; idx += 256) {
    const int i = idx >> 6, k = idx & 63;
    float m = -3.0e38f;
    for (int j = 0; j < 64; ++j) m = fmaxf(m, A2[i][j] + A1[j][k]);
    float s = 0.0f;
    for (int j = 0; j < 64; ++j) s += __expf(A2[i][j] + A1[j][k] - m);
    out[idx] = m + __logf(s);
  }
}

// ---------------------------------------------------------------------------
// Kernel 4: gold-path partial sums (deterministic 2-stage reduction)
// ---------------------------------------------------------------------------
__global__ __launch_bounds__(256) void gold_partial(const float* __restrict__ feats,
                                                    const int* __restrict__ tags,
                                                    const float* __restrict__ trans,
                                                    float* __restrict__ partials) {
  __shared__ float red[256];
  float s = 0.0f;
  for (int t = blockIdx.x * 256 + threadIdx.x; t < SEQ; t += 64 * 256) {
    const int cur  = tags[t];
    const int prev = (t == 0) ? START_ : tags[t - 1];
    s += feats[(size_t)t * KP + cur] + trans[cur * KTAG + prev];
  }
  red[threadIdx.x] = s;
  __syncthreads();
  for (int off = 128; off; off >>= 1) {
    if (threadIdx.x < off) red[threadIdx.x] += red[threadIdx.x + off];
    __syncthreads();
  }
  if (threadIdx.x == 0) partials[blockIdx.x] = red[0];
}

// ---------------------------------------------------------------------------
// Kernel 5: finalize. alpha_final = total[:, START]; forward = lse(alpha + T[STOP,:]);
// out = forward - gold.
// ---------------------------------------------------------------------------
__global__ __launch_bounds__(64) void crf_final(const float* __restrict__ total,
                                                const float* __restrict__ trans,
                                                const int* __restrict__ tags,
                                                const float* __restrict__ partials,
                                                float* __restrict__ out) {
  __shared__ float red[64];
  const int i = threadIdx.x;
  float v = -3.0e38f;
  if (i < KTAG) v = total[i * 64 + START_] + trans[STOP_ * KTAG + i];
  red[i] = v;
  __syncthreads();
  for (int off = 32; off; off >>= 1) {
    if (i < off) red[i] = fmaxf(red[i], red[i + off]);
    __syncthreads();
  }
  const float m = red[0];
  __syncthreads();
  red[i] = (i < KTAG) ? __expf(v - m) : 0.0f;
  __syncthreads();
  for (int off = 32; off; off >>= 1) {
    if (i < off) red[i] += red[i + off];
    __syncthreads();
  }
  if (i == 0) {
    const float fwd = m + __logf(red[0]);
    float gold = trans[STOP_ * KTAG + tags[SEQ - 1]];
    for (int bn = 0; bn < 64; ++bn) gold += partials[bn];
    out[0] = fwd - gold;
  }
}

// ---------------------------------------------------------------------------
extern "C" void kernel_launch(void* const* d_in, const int* in_sizes, int n_in,
                              void* d_out, int out_size, void* d_ws, size_t ws_size,
                              hipStream_t stream) {
  (void)in_sizes; (void)n_in; (void)out_size; (void)ws_size;
  const float* X     = (const float*)d_in[0];
  const int*   tags  = (const int*)d_in[1];
  const float* W     = (const float*)d_in[2];
  const float* bias  = (const float*)d_in[3];
  const float* trans = (const float*)d_in[4];
  float* out = (float*)d_out;

  float* ws       = (float*)d_ws;
  float* feats    = ws;                                   // 16384*64     = 1.05M f
  float* bufA     = feats + (size_t)SEQ * KP;             // 256*4096     = 1.05M f
  float* bufB     = bufA + (size_t)NCHUNK * 4096;         // 128*4096     = 0.52M f
  float* partials = bufB + (size_t)(NCHUNK / 2) * 4096;   // 64 f  (~10.5 MB total)

  feats_gemm<<<SEQ / 64, 128, 0, stream>>>(X, W, bias, feats);
  crf_chunk<<<NCHUNK, 128, 0, stream>>>(feats, trans, bufA);

  float* src = bufA;
  float* dst = bufB;
  for (int n = NCHUNK; n > 1; n >>= 1) {
    lse_combine<<<n / 2, 256, 0, stream>>>(src, dst);
    float* tmp = src; src = dst; dst = tmp;
  }

  gold_partial<<<64, 256, 0, stream>>>(feats, tags, trans, partials);
  crf_final<<<1, 64, 0, stream>>>(src, trans, tags, partials, out);
}
